// Node_classification_2_45492293599842
// MI455X (gfx1250) — matrix operand order repaired and checked by reading, not compiled, not run on hardware
//
#include <hip/hip_runtime.h>
#include <math.h>
#include <stdint.h>

// ---------------- problem constants ----------------
#define NB     64
#define LS     256
#define LAS    8
#define EMBED  300
#define EPAD   320          // EMBED padded to multiple of 32 for WMMA K
#define HID    384
#define DD     768          // 2*HID
#define KTOT   704          // EPAD + HID (fused [x | h] K dimension)
#define FINAL  768
#define NCLS   3
#define COLS_PER_WG 32      // hidden columns owned per workgroup
#define NWG    24           // 2 dirs * (HID/COLS_PER_WG)

typedef _Float16 half8 __attribute__((ext_vector_type(8)));
typedef _Float16 v16h  __attribute__((ext_vector_type(16)));
typedef float    v8f   __attribute__((ext_vector_type(8)));

// Branch-free sigmoid/tanh using v_exp_f32 + v_rcp_f32 (no IEEE div expansion,
// no EXEC-mask divergence on the recurrence critical path).
__device__ __forceinline__ float sigm(float x) {
    return __builtin_amdgcn_rcpf(1.0f + __expf(-x));
}
__device__ __forceinline__ float tanh_fast(float x) {
    // tanh(x) = 2*sigmoid(2x) - 1
    return 2.0f * __builtin_amdgcn_rcpf(1.0f + __expf(-2.0f * x)) - 1.0f;
}

// Async global -> LDS 16B copy (CDNA5 GLOBAL_LOAD_ASYNC_TO_LDS_B128, ASYNCcnt-tracked).
// LDS dest address = low 32 bits of the generic pointer (ISA 10.2: LDS aperture truncates
// to addr[31:0] = wave-relative LDS offset).
__device__ __forceinline__ void async_copy16(void* lds_dst, const void* gsrc) {
    uint32_t lo = (uint32_t)(uintptr_t)lds_dst;
    asm volatile("global_load_async_to_lds_b128 %0, %1, off"
                 :: "v"(lo), "v"(gsrc) : "memory");
}
__device__ __forceinline__ void wait_async0() {
    asm volatile("s_wait_asynccnt 0x0" ::: "memory");
}

// Build a 16-half WMMA fragment from two 8-half contiguous LDS chunks.
__device__ __forceinline__ v16h ld_frag(const _Float16* p0, const _Float16* p1) {
    half8 a = *(const half8*)p0;
    half8 b = *(const half8*)p1;
    v16h v;
#pragma unroll
    for (int i = 0; i < 8; ++i) { v[i] = a[i]; v[i + 8] = b[i]; }
    return v;
}

// Sense-reversing grid barrier over NWG persistent workgroups.
__device__ __forceinline__ void grid_sync(int* bar, int nwg) {
    __threadfence();
    __syncthreads();
    if (threadIdx.x == 0) {
        int gen = __hip_atomic_load(&bar[1], __ATOMIC_RELAXED, __HIP_MEMORY_SCOPE_AGENT);
        int t   = __hip_atomic_fetch_add(&bar[0], 1, __ATOMIC_ACQ_REL, __HIP_MEMORY_SCOPE_AGENT);
        if (t == nwg - 1) {
            __hip_atomic_store(&bar[0], 0, __ATOMIC_RELAXED, __HIP_MEMORY_SCOPE_AGENT);
            __hip_atomic_fetch_add(&bar[1], 1, __ATOMIC_ACQ_REL, __HIP_MEMORY_SCOPE_AGENT);
        } else {
            while (__hip_atomic_load(&bar[1], __ATOMIC_ACQUIRE, __HIP_MEMORY_SCOPE_AGENT) == gen)
                __builtin_amdgcn_s_sleep(2);
        }
    }
    __syncthreads();
}

__global__ void init_bar(int* bar) { bar[0] = 0; bar[1] = 0; }

// Embedding gather + f32->f16 convert, K padded 300->320 with zeros.
__global__ void gather_embed(const int* __restrict__ tok, const float* __restrict__ table,
                             _Float16* __restrict__ out, int rows) {
    int total = rows * EPAD;
    for (int i = blockIdx.x * blockDim.x + threadIdx.x; i < total; i += gridDim.x * blockDim.x) {
        int r = i / EPAD, k = i - r * EPAD;
        float v = (k < EMBED) ? table[(size_t)tok[r] * EMBED + k] : 0.0f;
        out[i] = (_Float16)v;
    }
}

// Persistent fused BiLSTM. 24 WGs: dir = blockIdx&1, column block = blockIdx>>1.
// LDS: weight slice [4 gates][32 cols][KTOT] f16 (176KB) + activations [64][KTOT] f16 (88KB).
__global__ void lstm_persistent(
    const _Float16* __restrict__ X, int Lcur,
    const float* __restrict__ wih_f, const float* __restrict__ whh_f,
    const float* __restrict__ bih_f, const float* __restrict__ bhh_f,
    const float* __restrict__ wih_b, const float* __restrict__ whh_b,
    const float* __restrict__ bih_b, const float* __restrict__ bhh_b,
    _Float16* __restrict__ hbuf,          // [2][NB][HID] f16
    float* __restrict__ feat,             // [NB][LS][DD] (sentence mode)
    float* __restrict__ asp_sum,          // [NB][DD]     (aspect mode)
    int is_aspect, int* __restrict__ bar)
{
    extern __shared__ _Float16 smem[];
    _Float16* Ws = smem;                               // [4*COLS_PER_WG][KTOT]
    _Float16* As = smem + 4 * COLS_PER_WG * KTOT;      // [NB][KTOT]

    const int tid = threadIdx.x;
    const int dir = blockIdx.x & 1;
    const int cb  = (blockIdx.x >> 1) * COLS_PER_WG;

    const float* wih = dir ? wih_b : wih_f;
    const float* whh = dir ? whh_b : whh_f;
    const float* bih = dir ? bih_b : bih_f;
    const float* bhh = dir ? bhh_b : bhh_f;
    _Float16* hb = hbuf + dir * NB * HID;

    // Stage weight slice into LDS once (f32 -> f16), rows = gate*32 + localCol.
    for (int idx = tid; idx < 4 * COLS_PER_WG * KTOT; idx += blockDim.x) {
        int r = idx / KTOT, k = idx - r * KTOT;
        int g = r >> 5, lc = r & 31;
        int orow = g * HID + cb + lc;
        float v;
        if (k < EPAD) v = (k < EMBED) ? wih[orow * EMBED + k] : 0.0f;
        else          v = whh[orow * HID + (k - EPAD)];
        Ws[idx] = (_Float16)v;
    }
    // h0 = 0 for our columns.
    for (int idx = tid; idx < NB * COLS_PER_WG; idx += blockDim.x) {
        int m = idx >> 5, c = idx & 31;
        hb[m * HID + cb + c] = (_Float16)0.0f;
    }

    const int lane  = tid & 31;
    const int wave  = tid >> 5;
    const int mtile = wave & 3;     // M tiles of 16 over 64 batch rows
    const int nh    = wave >> 2;    // 2 tiles of 16 hidden cols

    const int nloc = nh * 16 + (lane & 15);
    const int col  = cb + nloc;
    const float bi_i = bih[0 * HID + col] + bhh[0 * HID + col];
    const float bi_f = bih[1 * HID + col] + bhh[1 * HID + col];
    const float bi_g = bih[2 * HID + col] + bhh[2 * HID + col];
    const float bi_o = bih[3 * HID + col] + bhh[3 * HID + col];

    float creg[8], hsum[8];
#pragma unroll
    for (int r = 0; r < 8; ++r) { creg[r] = 0.0f; hsum[r] = 0.0f; }

    grid_sync(bar, NWG);   // weights staged + h zeroed everywhere

    const int arow  = mtile * 16 + (lane & 15);
    const int akoff = (lane < 16) ? 0 : 16;    // A-fragment K base (contiguous k = 16h + s)
    const int bkoff = (lane < 16) ? 0 : 16;    // B-fragment K base (contiguous k = 16h + s)

    for (int step = 0; step < Lcur; ++step) {
        const int t = dir ? (Lcur - 1 - step) : step;

        // Stage As = [x_t | h_{t-1}] via async direct-to-LDS b128 copies.
        for (int q = tid; q < NB * (EPAD / 8); q += blockDim.x) {
            int m = q / (EPAD / 8), kc = (q - m * (EPAD / 8)) * 8;
            async_copy16(&As[m * KTOT + kc], &X[((size_t)m * Lcur + t) * EPAD + kc]);
        }
        for (int q = tid; q < NB * (HID / 8); q += blockDim.x) {
            int m = q / (HID / 8), kc = (q - m * (HID / 8)) * 8;
            async_copy16(&As[m * KTOT + EPAD + kc], &hb[m * HID + kc]);
        }
        wait_async0();
        __syncthreads();

        v8f acc0 = {}, acc1 = {}, acc2 = {}, acc3 = {};
#pragma unroll 2
        for (int kt = 0; kt < KTOT / 32; ++kt) {
            const int k0 = kt * 32;
            v16h av = ld_frag(&As[arow * KTOT + k0 + akoff],
                              &As[arow * KTOT + k0 + akoff + 8]);
            v16h b0 = ld_frag(&Ws[(0 * COLS_PER_WG + nloc) * KTOT + k0 + bkoff],
                              &Ws[(0 * COLS_PER_WG + nloc) * KTOT + k0 + bkoff + 8]);
            acc0 = __builtin_amdgcn_wmma_f32_16x16x32_f16(false, av, false, b0, (short)0, acc0, false, false);
            v16h b1 = ld_frag(&Ws[(1 * COLS_PER_WG + nloc) * KTOT + k0 + bkoff],
                              &Ws[(1 * COLS_PER_WG + nloc) * KTOT + k0 + bkoff + 8]);
            acc1 = __builtin_amdgcn_wmma_f32_16x16x32_f16(false, av, false, b1, (short)0, acc1, false, false);
            v16h b2 = ld_frag(&Ws[(2 * COLS_PER_WG + nloc) * KTOT + k0 + bkoff],
                              &Ws[(2 * COLS_PER_WG + nloc) * KTOT + k0 + bkoff + 8]);
            acc2 = __builtin_amdgcn_wmma_f32_16x16x32_f16(false, av, false, b2, (short)0, acc2, false, false);
            v16h b3 = ld_frag(&Ws[(3 * COLS_PER_WG + nloc) * KTOT + k0 + bkoff],
                              &Ws[(3 * COLS_PER_WG + nloc) * KTOT + k0 + bkoff + 8]);
            acc3 = __builtin_amdgcn_wmma_f32_16x16x32_f16(false, av, false, b3, (short)0, acc3, false, false);
        }

        // LSTM cell, pytorch gate order (i,f,g,o); c lives in regs across steps.
#pragma unroll
        for (int r = 0; r < 8; ++r) {
            float iv = acc0[r] + bi_i;
            float fv = acc1[r] + bi_f;
            float gv = acc2[r] + bi_g;
            float ov = acc3[r] + bi_o;
            float cn = sigm(fv) * creg[r] + sigm(iv) * tanh_fast(gv);
            creg[r] = cn;
            float hv = sigm(ov) * tanh_fast(cn);
            int m = mtile * 16 + r + ((lane >> 4) << 3);   // C/D VGPR layout
            hb[m * HID + col] = (_Float16)hv;
            if (is_aspect) hsum[r] += hv;
            else feat[((size_t)m * LS + t) * DD + dir * HID + col] = hv;
        }
        grid_sync(bar, NWG);   // publish h_t before anyone stages step t+1
    }

    if (is_aspect) {
#pragma unroll
        for (int r = 0; r < 8; ++r) {
            int m = mtile * 16 + r + ((lane >> 4) << 3);
            asp_sum[m * DD + dir * HID + col] = hsum[r];
        }
    }
}

// Masked-softmax attention + pooled = sum(feat*att)/NUM_HEADS. One block per batch row.
__global__ void attention_pool(const float* __restrict__ feat, const float* __restrict__ asp_sum,
                               const int* __restrict__ text_len, float* __restrict__ pooled) {
    __shared__ float aspv[DD];
    __shared__ float att[LS];
    __shared__ float red[LS];
    const int n = blockIdx.x, tid = threadIdx.x;
    for (int d = tid; d < DD; d += blockDim.x) aspv[d] = asp_sum[n * DD + d] * (1.0f / LAS);
    __syncthreads();
    const int tl = text_len[n];
    {
        const float* fr = feat + ((size_t)n * LS + tid) * DD;
        float dv = 0.0f;
        for (int d = 0; d < DD; ++d) dv += fr[d] * aspv[d];
        float mval = (tid < tl) ? dv : -1e30f;
        att[tid] = mval; red[tid] = mval;
    }
    __syncthreads();
    for (int s = LS / 2; s > 0; s >>= 1) {
        if (tid < s) red[tid] = fmaxf(red[tid], red[tid + s]);
        __syncthreads();
    }
    const float mx = red[0];
    __syncthreads();
    { float e = __expf(att[tid] - mx); att[tid] = e; red[tid] = e; }
    __syncthreads();
    for (int s = LS / 2; s > 0; s >>= 1) {
        if (tid < s) red[tid] += red[tid + s];
        __syncthreads();
    }
    const float inv = 1.0f / red[0];
    __syncthreads();
    att[tid] *= inv;
    __syncthreads();
    for (int d = tid; d < DD; d += blockDim.x) {
        float s = 0.0f;
        for (int l = 0; l < LS; ++l) s += feat[((size_t)n * LS + l) * DD + d] * att[l];
        pooled[n * DD + d] = s * (1.0f / 6.0f);
    }
}

// Tiny fp32 GEMM: Y[n,:] = act(X[n,:] @ W^T + b). One block per batch row.
__global__ void mlp_kernel(const float* __restrict__ X, const float* __restrict__ W,
                           const float* __restrict__ b, float* __restrict__ Y,
                           int K, int Nout, int doRelu) {
    __shared__ float xs[FINAL];
    const int n = blockIdx.x;
    for (int k = threadIdx.x; k < K; k += blockDim.x) xs[k] = X[n * K + k];
    __syncthreads();
    for (int j = threadIdx.x; j < Nout; j += blockDim.x) {
        float acc = b[j];
        const float* wr = W + (size_t)j * K;
        for (int k = 0; k < K; ++k) acc += xs[k] * wr[k];
        if (doRelu) acc = fmaxf(acc, 0.0f);
        Y[n * Nout + j] = acc;
    }
}

// d_out = [node_logit (N,L,3) | node_feature_final (N,L,768)], both broadcast over L.
__global__ void broadcast_out(const float* __restrict__ pooled, const float* __restrict__ logits,
                              float* __restrict__ out) {
    const int T1 = NB * LS * NCLS;
    const int T2 = NB * LS * DD;
    float* outL = out;
    float* outF = out + T1;
    for (int i = blockIdx.x * blockDim.x + threadIdx.x; i < T2; i += gridDim.x * blockDim.x) {
        int n = i / (LS * DD);
        int d = i % DD;
        outF[i] = pooled[n * DD + d];
        if (i < T1) {
            int nn = i / (LS * NCLS);
            int c  = i % NCLS;
            outL[i] = logits[nn * NCLS + c];
        }
    }
}

extern "C" void kernel_launch(void* const* d_in, const int* in_sizes, int n_in,
                              void* d_out, int out_size, void* d_ws, size_t ws_size,
                              hipStream_t stream) {
    (void)in_sizes; (void)n_in; (void)out_size; (void)ws_size;
    const int*   sentence = (const int*)d_in[0];
    const int*   aspect   = (const int*)d_in[1];
    const int*   text_len = (const int*)d_in[2];
    const float* table    = (const float*)d_in[3];
    const float* wih_f = (const float*)d_in[4];
    const float* whh_f = (const float*)d_in[5];
    const float* bih_f = (const float*)d_in[6];
    const float* bhh_f = (const float*)d_in[7];
    const float* wih_b = (const float*)d_in[8];
    const float* whh_b = (const float*)d_in[9];
    const float* bih_b = (const float*)d_in[10];
    const float* bhh_b = (const float*)d_in[11];
    const float* W1 = (const float*)d_in[12];
    const float* b1 = (const float*)d_in[13];
    const float* W2 = (const float*)d_in[14];
    const float* b2 = (const float*)d_in[15];
    const float* Wf = (const float*)d_in[16];
    const float* bf = (const float*)d_in[17];

    char* ws = (char*)d_ws;
    size_t off = 0;
    auto take = [&](size_t bytes) -> char* {
        char* p = ws + off;
        off = (off + bytes + 255) & ~(size_t)255;
        return p;
    };
    _Float16* Xs      = (_Float16*)take((size_t)NB * LS * EPAD * sizeof(_Float16));
    _Float16* Xa      = (_Float16*)take((size_t)NB * LAS * EPAD * sizeof(_Float16));
    float*    feat    = (float*)take((size_t)NB * LS * DD * sizeof(float));
    _Float16* hbuf    = (_Float16*)take((size_t)2 * NB * HID * sizeof(_Float16));
    float*    asp_sum = (float*)take((size_t)NB * DD * sizeof(float));
    float*    pooled  = (float*)take((size_t)NB * DD * sizeof(float));
    float*    x1      = (float*)take((size_t)NB * FINAL * sizeof(float));
    float*    x2      = (float*)take((size_t)NB * FINAL * sizeof(float));
    float*    logits  = (float*)take((size_t)NB * NCLS * sizeof(float));
    int*      bar     = (int*)take(256);

    hipLaunchKernelGGL(init_bar, dim3(1), dim3(1), 0, stream, bar);
    hipLaunchKernelGGL(gather_embed, dim3(2048), dim3(256), 0, stream, sentence, table, Xs, NB * LS);
    hipLaunchKernelGGL(gather_embed, dim3(128), dim3(256), 0, stream, aspect, table, Xa, NB * LAS);

    const size_t smem = (size_t)(4 * COLS_PER_WG * KTOT + NB * KTOT) * sizeof(_Float16);
    hipLaunchKernelGGL(lstm_persistent, dim3(NWG), dim3(256), smem, stream,
                       Xs, LS, wih_f, whh_f, bih_f, bhh_f, wih_b, whh_b, bih_b, bhh_b,
                       hbuf, feat, (float*)nullptr, 0, bar);
    hipLaunchKernelGGL(lstm_persistent, dim3(NWG), dim3(256), smem, stream,
                       Xa, LAS, wih_f, whh_f, bih_f, bhh_f, wih_b, whh_b, bih_b, bhh_b,
                       hbuf, (float*)nullptr, asp_sum, 1, bar);

    hipLaunchKernelGGL(attention_pool, dim3(NB), dim3(LS), 0, stream, feat, asp_sum, text_len, pooled);
    hipLaunchKernelGGL(mlp_kernel, dim3(NB), dim3(256), 0, stream, pooled, W1, b1, x1, FINAL, FINAL, 1);
    hipLaunchKernelGGL(mlp_kernel, dim3(NB), dim3(256), 0, stream, x1, W2, b2, x2, FINAL, FINAL, 1);
    hipLaunchKernelGGL(mlp_kernel, dim3(NB), dim3(256), 0, stream, x2, Wf, bf, logits, FINAL, NCLS, 0);
    hipLaunchKernelGGL(broadcast_out, dim3(4096), dim3(256), 0, stream, pooled, logits, (float*)d_out);
}
